// SegmentCausalCrossAttentionFlex_14456859919083
// MI455X (gfx1250) — compile-verified, matrix operand order checked by reading, not run on hardware
//
#include <hip/hip_runtime.h>
#include <hip/hip_bf16.h>

// ---------------------------------------------------------------------------
// SegmentCausalCrossAttentionFlex for MI455X (gfx1250, wave32, WMMA)
//
//   pack:  q,kv,ctx -> f16 ; Wq,Wkv,Wo -> f16 transposed [N][K]
//   qh  = rope(q @ Wq)          [B*LQ, 1024]   (WMMA f16 GEMM, async LDS)
//   kvl = kv @ Wkv              [B*LK, 2048]   (WMMA f16 GEMM, async LDS)
//   ctx = window-attn(qh, kh, vh, seg)         (<=5 keys/query, scalar)
//   out = ctx @ Wo              [B*LQ, 1024]   (WMMA f16 GEMM, async LDS)
// ---------------------------------------------------------------------------

typedef __attribute__((ext_vector_type(16))) _Float16 v16h;
typedef __attribute__((ext_vector_type(8)))  _Float16 v8h;
typedef __attribute__((ext_vector_type(8)))  float    v8f;
typedef int v4i_vs __attribute__((vector_size(4 * sizeof(int))));

#define BATCH    2
#define LQ       4096
#define LK       512
#define DMODEL   1024
#define NHEADS   16
#define HDIM     64
#define LOOKBACK 4
#define ROPE_MAX 8192

// ---- CDNA5 async global->LDS copy (ASYNCcnt), guarded with fallback ----
#if defined(__has_builtin)
#if __has_builtin(__builtin_amdgcn_global_load_async_to_lds_b128)
#define HAVE_ASYNC_LDS 1
#endif
#endif

#if HAVE_ASYNC_LDS
__device__ __forceinline__ void async_cp16(const _Float16* gsrc, _Float16* lds)
{
    // b128 async copy: 16B from global to LDS, tracked by ASYNCcnt
    __attribute__((address_space(1))) v4i_vs* g =
        (__attribute__((address_space(1))) v4i_vs*)const_cast<_Float16*>(gsrc);
    __attribute__((address_space(3))) v4i_vs* l =
        (__attribute__((address_space(3))) v4i_vs*)lds;
    __builtin_amdgcn_global_load_async_to_lds_b128(g, l, 0, 0);
}
#endif

__device__ __forceinline__ void wait_asynccnt0() {
#if defined(__has_builtin) && __has_builtin(__builtin_amdgcn_s_wait_asynccnt)
    __builtin_amdgcn_s_wait_asynccnt(0);
#else
    asm volatile("s_wait_asynccnt 0x0" ::: "memory");
#endif
}

// ============================================================================
// Tiled WMMA GEMM: C[M,N] = A[M,K] * Bt[N,K]^T, f16 in / f32 out.
// Block: 256 threads (8 waves). Tile 128x128, BK=32, double-buffered LDS.
// Wave w owns rows [16w,16w+16) x 128 cols -> 8 accumulators, 8 WMMA/k-step.
// ============================================================================
#define TM 128
#define TN 128
#define TK 32
#define LDS_PAD 40   // halves per LDS row (80B: bank-skewed, 16B aligned)

template <int N, int K>
__global__ __launch_bounds__(256)
void gemm_f16t_wmma_kernel(const _Float16* __restrict__ A,
                           const _Float16* __restrict__ Bt,
                           float* __restrict__ C, int M)
{
    __shared__ __attribute__((aligned(16))) _Float16 As[2][TM * LDS_PAD];
    __shared__ __attribute__((aligned(16))) _Float16 Bs[2][TN * LDS_PAD];

    const int tid   = threadIdx.x;
    const int lane  = tid & 31;
    const int wave  = tid >> 5;
    const int tileM = blockIdx.x * TM;
    const int tileN = blockIdx.y * TN;

    v8f acc[8] = {};

    // fragment addressing (ISA 16-bit A/B layout):
    // lane<16 -> K {0..7,16..23}; lane>=16 -> K {8..15,24..31}
    const int frow = lane & 15;
    const int koff = (lane >= 16) ? 8 : 0;

    // staging: thread copies 32B of A row and 32B of Bt row per tile
    const int sRow = tid >> 1;          // 0..127
    const int sOff = (tid & 1) * 16;    // halves: 0 or 16
    const _Float16* Ag = A  + (size_t)(tileM + sRow) * K + sOff;
    const _Float16* Bg = Bt + (size_t)(tileN + sRow) * K + sOff;
    const int sLds = sRow * LDS_PAD + sOff;

    auto stage = [&](int kt, int buf) {
#if HAVE_ASYNC_LDS
        async_cp16(Ag + kt,     &As[buf][sLds]);
        async_cp16(Ag + kt + 8, &As[buf][sLds + 8]);
        async_cp16(Bg + kt,     &Bs[buf][sLds]);
        async_cp16(Bg + kt + 8, &Bs[buf][sLds + 8]);
#else
        const v8h a0 = *(const v8h*)(Ag + kt);
        const v8h a1 = *(const v8h*)(Ag + kt + 8);
        const v8h b0 = *(const v8h*)(Bg + kt);
        const v8h b1 = *(const v8h*)(Bg + kt + 8);
        *(v8h*)&As[buf][sLds]     = a0;
        *(v8h*)&As[buf][sLds + 8] = a1;
        *(v8h*)&Bs[buf][sLds]     = b0;
        *(v8h*)&Bs[buf][sLds + 8] = b1;
#endif
    };

    auto sync_stage = [&]() {
#if HAVE_ASYNC_LDS
        wait_asynccnt0();
#endif
        __syncthreads();
    };

    auto compute = [&](int buf) {
        // A fragment (2x ds_load_b128)
        v16h afrag;
        {
            const _Float16* base = &As[buf][(16 * wave + frow) * LDS_PAD];
            const v8h lo = *(const v8h*)(base + koff);
            const v8h hi = *(const v8h*)(base + 16 + koff);
            #pragma unroll
            for (int i = 0; i < 8; ++i) { afrag[i] = lo[i]; afrag[i + 8] = hi[i]; }
        }
        // 8 B fragments first (clause of ds_load_b128), then 8 WMMAs
        v16h bfrag[8];
        #pragma unroll
        for (int nb = 0; nb < 8; ++nb) {
            const _Float16* base = &Bs[buf][(nb * 16 + frow) * LDS_PAD];
            const v8h lo = *(const v8h*)(base + koff);
            const v8h hi = *(const v8h*)(base + 16 + koff);
            #pragma unroll
            for (int i = 0; i < 8; ++i) { bfrag[nb][i] = lo[i]; bfrag[nb][i + 8] = hi[i]; }
        }
        #pragma unroll
        for (int nb = 0; nb < 8; ++nb) {
            acc[nb] = __builtin_amdgcn_wmma_f32_16x16x32_f16(
                          false, afrag, false, bfrag[nb], (short)0, acc[nb], false, false);
        }
#if defined(__has_builtin) && __has_builtin(__builtin_amdgcn_sched_group_barrier)
        // keep all fragments in flight: 18 DS-reads first, then 8 WMMAs
        __builtin_amdgcn_sched_group_barrier(0x100, 18, 0);  // DS read group
        __builtin_amdgcn_sched_group_barrier(0x008, 8, 0);   // MFMA/WMMA group
#endif
    };

    // ---- double-buffered pipeline: 1 barrier per k-step ----
    stage(0, 0);
    for (int kt = 0; kt < K; kt += 2 * TK) {
        sync_stage();
        if (kt + TK < K) stage(kt + TK, 1);
        compute(0);
        sync_stage();
        if (kt + 2 * TK < K) stage(kt + 2 * TK, 0);
        compute(1);
    }

    // ---- C store: single base + compile-time immediate offsets ----
    // lane l -> col (l&15), rows (l>>4)*8 + r   (ISA 16x16 f32 C/D layout)
    const int nlocal = lane & 15;
    const int mhalf  = (lane >> 4) * 8;
    float* Cb = C + (size_t)(tileM + 16 * wave + mhalf) * N + tileN + nlocal;
    #pragma unroll
    for (int nb = 0; nb < 8; ++nb) {
        #pragma unroll
        for (int r = 0; r < 8; ++r) {
            Cb[r * N + nb * 16] = acc[nb][r];
        }
    }
}

// ---- one-time pack: fp32 -> f16, straight copy (n multiple of 8) ----
__global__ __launch_bounds__(256)
void pack_f16_kernel(const float* __restrict__ in, _Float16* __restrict__ out, int n)
{
    const int i = (blockIdx.x * blockDim.x + threadIdx.x) * 8;
    if (i >= n) return;
    const float4 a = *(const float4*)(in + i);
    const float4 b = *(const float4*)(in + i + 4);
    v8h v;
    v[0] = (_Float16)a.x; v[1] = (_Float16)a.y; v[2] = (_Float16)a.z; v[3] = (_Float16)a.w;
    v[4] = (_Float16)b.x; v[5] = (_Float16)b.y; v[6] = (_Float16)b.z; v[7] = (_Float16)b.w;
    *(v8h*)(out + i) = v;
}

// ---- one-time pack: fp32 [K,N] -> f16 transposed [N,K] (LDS-tiled) ----
__global__ __launch_bounds__(256)
void pack_f16_transpose_kernel(const float* __restrict__ in, _Float16* __restrict__ out,
                               int K, int N)
{
    __shared__ float tile[32][33];
    const int k0 = blockIdx.y * 32;
    const int n0 = blockIdx.x * 32;
    const int tx = threadIdx.x;              // 0..31
    for (int i = threadIdx.y; i < 32; i += 8)
        tile[i][tx] = in[(size_t)(k0 + i) * N + n0 + tx];
    __syncthreads();
    for (int i = threadIdx.y; i < 32; i += 8)
        out[(size_t)(n0 + i) * K + k0 + tx] = (_Float16)tile[tx][i];
}

// ---- in-place gathered RoPE on [nrows, ..., NHEADS*HDIM] slices ----
__global__ __launch_bounds__(256)
void rope_kernel(float* __restrict__ x, const int* __restrict__ pos,
                 int nrows, int rowStride, int posLen)
{
    const int idx   = blockIdx.x * blockDim.x + threadIdx.x;
    const int total = nrows * NHEADS * 32;
    if (idx >= total) return;
    const int i   = idx & 31;
    const int h   = (idx >> 5) & (NHEADS - 1);
    const int row = idx >> 9;

    int p = pos[row % posLen];
    p = min(max(p, 0), ROPE_MAX - 1);

    const float inv_freq = __powf(10000.0f, -(float)i / 32.0f);
    const float f = (float)p * inv_freq;
    const float c = __cosf(f);
    const float s = __sinf(f);

    float* base = x + (size_t)row * rowStride + h * HDIM;
    const float x0 = base[i];
    const float x1 = base[i + 32];
    base[i]      = x0 * c - x1 * s;
    base[i + 32] = x1 * c + x0 * s;
}

// ---- windowed attention: <=5 keys per query, one thread per (b,q,h) ----
__global__ __launch_bounds__(256)
void window_attn_kernel(const float* __restrict__ qp, const float* __restrict__ kvl,
                        const int* __restrict__ seg, float* __restrict__ ctx)
{
    const int idx = blockIdx.x * blockDim.x + threadIdx.x;
    if (idx >= BATCH * LQ * NHEADS) return;
    const int h  = idx & (NHEADS - 1);
    const int qi = (idx >> 4) & (LQ - 1);
    const int b  = idx >> 16;

    const int s   = seg[b * LQ + qi];
    const int jlo = (s - LOOKBACK > 0) ? (s - LOOKBACK) : 0;
    const int nk  = s - jlo + 1;

    const float* qv = qp + ((size_t)(b * LQ + qi)) * DMODEL + h * HDIM;
    float qreg[HDIM];
    #pragma unroll
    for (int d = 0; d < HDIM; ++d) qreg[d] = qv[d];

    float sc[LOOKBACK + 1];
    float mmax = -3.4e38f;
    for (int jj = 0; jj < nk; ++jj) {
        const float* kvp = kvl + ((size_t)(b * LK + jlo + jj)) * (2 * DMODEL) + h * HDIM;
        float dot = 0.0f;
        #pragma unroll
        for (int d = 0; d < HDIM; ++d) dot += qreg[d] * kvp[d];
        dot *= 0.125f;                       // 1/sqrt(64)
        sc[jj] = dot;
        mmax = fmaxf(mmax, dot);
    }
    float denom = 0.0f;
    for (int jj = 0; jj < nk; ++jj) {
        sc[jj] = __expf(sc[jj] - mmax);
        denom += sc[jj];
    }
    const float rdenom = 1.0f / denom;

    float out[HDIM];
    #pragma unroll
    for (int d = 0; d < HDIM; ++d) out[d] = 0.0f;
    for (int jj = 0; jj < nk; ++jj) {
        const float w = sc[jj] * rdenom;
        const float* vv = kvl + ((size_t)(b * LK + jlo + jj)) * (2 * DMODEL)
                              + DMODEL + h * HDIM;
        #pragma unroll
        for (int d = 0; d < HDIM; ++d) out[d] += w * vv[d];
    }

    float* cp = ctx + ((size_t)(b * LQ + qi)) * DMODEL + h * HDIM;
    #pragma unroll
    for (int d = 0; d < HDIM; ++d) cp[d] = out[d];
}

extern "C" void kernel_launch(void* const* d_in, const int* in_sizes, int n_in,
                              void* d_out, int out_size, void* d_ws, size_t ws_size,
                              hipStream_t stream) {
    const float* q      = (const float*)d_in[0];   // [B, LQ, 1024]
    const float* kv     = (const float*)d_in[1];   // [B, LK, 1024]
    const int*   q_pos  = (const int*)d_in[2];     // [B, LQ]
    const int*   kv_pos = (const int*)d_in[3];     // [LK]
    const int*   seg    = (const int*)d_in[4];     // [B, LQ]
    const float* Wq     = (const float*)d_in[5];   // [1024, 1024]
    const float* Wkv    = (const float*)d_in[6];   // [1024, 2048]
    const float* Wo     = (const float*)d_in[7];   // [1024, 1024]
    float*       out    = (float*)d_out;           // [B, LQ, 1024]

    const int Mq = BATCH * LQ;     // 8192
    const int Mk = BATCH * LK;     // 1024

    // ---- workspace layout ----
    char* ws = (char*)d_ws;
    size_t off = 0;
    auto alloc = [&](size_t bytes) { void* p = ws + off; off += (bytes + 255) & ~(size_t)255; return p; };
    float*    qproj  = (float*)   alloc((size_t)Mq * DMODEL * 4);      // f32
    float*    kvl    = (float*)   alloc((size_t)Mk * 2 * DMODEL * 4);  // f32
    float*    ctx    = (float*)   alloc((size_t)Mq * DMODEL * 4);      // f32
    _Float16* qf     = (_Float16*)alloc((size_t)Mq * DMODEL * 2);
    _Float16* kvf    = (_Float16*)alloc((size_t)Mk * DMODEL * 2);
    _Float16* ctxf   = (_Float16*)alloc((size_t)Mq * DMODEL * 2);
    _Float16* Wqt    = (_Float16*)alloc((size_t)DMODEL * DMODEL * 2);      // [N=1024][K=1024]
    _Float16* Wkvt   = (_Float16*)alloc((size_t)2 * DMODEL * DMODEL * 2);  // [N=2048][K=1024]
    _Float16* Wot    = (_Float16*)alloc((size_t)DMODEL * DMODEL * 2);

    // ---- 0) one-time packs (f16 activations, f16 transposed weights) ----
    {
        int n = Mq * DMODEL;
        pack_f16_kernel<<<(n / 8 + 255) / 256, 256, 0, stream>>>(q, qf, n);
        n = Mk * DMODEL;
        pack_f16_kernel<<<(n / 8 + 255) / 256, 256, 0, stream>>>(kv, kvf, n);
        pack_f16_transpose_kernel<<<dim3(DMODEL / 32, DMODEL / 32), dim3(32, 8), 0, stream>>>(
            Wq, Wqt, DMODEL, DMODEL);
        pack_f16_transpose_kernel<<<dim3(2 * DMODEL / 32, DMODEL / 32), dim3(32, 8), 0, stream>>>(
            Wkv, Wkvt, DMODEL, 2 * DMODEL);
        pack_f16_transpose_kernel<<<dim3(DMODEL / 32, DMODEL / 32), dim3(32, 8), 0, stream>>>(
            Wo, Wot, DMODEL, DMODEL);
    }

    // ---- 1) qproj = q @ Wq      [8192,1024]x[1024,1024] ----
    gemm_f16t_wmma_kernel<DMODEL, DMODEL>
        <<<dim3(Mq / TM, DMODEL / TN), 256, 0, stream>>>(qf, Wqt, qproj, Mq);

    // ---- 2) kvl = kv @ Wkv      [1024,1024]x[1024,2048] ----
    gemm_f16t_wmma_kernel<2 * DMODEL, DMODEL>
        <<<dim3(Mk / TM, (2 * DMODEL) / TN), 256, 0, stream>>>(kvf, Wkvt, kvl, Mk);

    // ---- 3) RoPE in place ----
    {
        int total = Mq * NHEADS * 32;
        rope_kernel<<<(total + 255) / 256, 256, 0, stream>>>(qproj, q_pos, Mq, DMODEL, Mq);
        total = Mk * NHEADS * 32;
        rope_kernel<<<(total + 255) / 256, 256, 0, stream>>>(kvl, kv_pos, Mk, 2 * DMODEL, LK);
    }

    // ---- 4) windowed attention (<=5 keys/query) ----
    {
        const int total = BATCH * LQ * NHEADS;
        window_attn_kernel<<<(total + 255) / 256, 256, 0, stream>>>(qproj, kvl, seg, ctx);
    }

    // ---- 5) out = ctx @ Wo      [8192,1024]x[1024,1024] ----
    {
        const int n = Mq * DMODEL;
        pack_f16_kernel<<<(n / 8 + 255) / 256, 256, 0, stream>>>(ctx, ctxf, n);
        gemm_f16t_wmma_kernel<DMODEL, DMODEL>
            <<<dim3(Mq / TM, DMODEL / TN), 256, 0, stream>>>(ctxf, Wot, out, Mq);
    }
}
